// TransformerConvNet_22935125360683
// MI455X (gfx1250) — matrix operand order repaired and checked by reading, not compiled, not run on hardware
//
#include <hip/hip_runtime.h>
#include <math.h>

// ---------------- CDNA5 WMMA types ----------------
typedef __attribute__((ext_vector_type(16))) _Float16 v16h;
typedef __attribute__((ext_vector_type(8)))  float    v8f;

// ---------------- helpers ----------------
__device__ __forceinline__ void atomicMaxFloat(float* addr, float value) {
    // Monotonic-bits trick: works with -inf init, no CAS loop.
    if (value >= 0.0f)
        atomicMax((int*)addr, __float_as_int(value));
    else
        atomicMin((unsigned int*)addr, __float_as_uint(value));
}

__global__ void fill_kernel(float* __restrict__ p, float v, int count) {
    int i = blockIdx.x * blockDim.x + threadIdx.x;
    if (i < count) p[i] = v;
}

// x0[n,16] = emb[node_idx[n], :]
__global__ void embed_kernel(const int* __restrict__ node_idx,
                             const float* __restrict__ emb,
                             float* __restrict__ x0, int N) {
    int i = blockIdx.x * blockDim.x + threadIdx.x;
    if (i >= N * 16) return;
    int n = i >> 4, k = i & 15;
    x0[i] = emb[node_idx[n] * 16 + k];
}

// ---------------- WMMA GEMM: Q/K/V/S = x @ W* + b* ----------------
// Block = 128 threads = 4 waves; wave w computes weight matrix w.
// Fin/F are compile-time; all guards are branchless (clamp + select) so the
// MMA pipeline is straight-line code with full EXEC, and the f16 A fragments
// are built once per wave from an LDS-staged zero-padded tile.
template<int Fin, int F>
__global__ void __launch_bounds__(128)
gemm4_wmma_kernel(const float* __restrict__ x, int N,
    const float* __restrict__ W0q, const float* __restrict__ W1k,
    const float* __restrict__ W2v, const float* __restrict__ W3s,
    const float* __restrict__ B0q, const float* __restrict__ B1k,
    const float* __restrict__ B2v, const float* __restrict__ B3s,
    float* __restrict__ O0q, float* __restrict__ O1k,
    float* __restrict__ O2v, float* __restrict__ O3s)
{
    constexpr int nK   = (Fin + 31) / 32;   // K steps of 32
    constexpr int KPAD = nK * 32;           // zero-padded K extent
    constexpr int nCT  = F / 16;            // 16-wide col tiles

    __shared__ _Float16 xs[16 * KPAD];

    const int row0 = blockIdx.x * 16;
    const int tid  = threadIdx.x;

    // Stage x tile (16 x Fin) as f16, zero-padded to KPAD / past N (branchless).
    #pragma unroll
    for (int t = tid; t < 16 * KPAD; t += 128) {
        const int r = t / KPAD, k = t - r * KPAD;
        const int row  = row0 + r;
        const int rowc = (row < N) ? row : (N - 1);
        const int kc   = (k < Fin) ? k : 0;
        float v = x[(unsigned)rowc * Fin + kc];
        v = (k < Fin && row < N) ? v : 0.0f;
        xs[t] = (_Float16)v;
    }
    __syncthreads();

    const int wave = tid >> 5;
    const int lane = tid & 31;
    const int g    = (lane >> 4) & 1;   // half-wave group
    const int nn   = lane & 15;         // col within tile / A row

    const float* W; const float* bias; float* Out;
    switch (wave) {
        case 0:  W = W0q; bias = B0q; Out = O0q; break;
        case 1:  W = W1k; bias = B1k; Out = O1k; break;
        case 2:  W = W2v; bias = B2v; Out = O2v; break;
        default: W = W3s; bias = B3s; Out = O3s; break;
    }

    __builtin_prefetch(W, 0, 1);            // global_prefetch_b8 -> prime GL2

    // A fragments (16x32 f16 per K-step, row m = nn), built once.
    // VGPR j (j<4): K = 8g+2j+{0,1};  (j>=4): K = 16+8g+2(j-4)+{0,1}.
    v16h A[nK];
    #pragma unroll
    for (int ks = 0; ks < nK; ++ks) {
        #pragma unroll
        for (int j = 0; j < 8; ++j) {
            const int kbA = (j < 4) ? (8 * g + 2 * j) : (16 + 8 * g + 2 * (j - 4));
            #pragma unroll
            for (int e = 0; e < 2; ++e)
                A[ks][2 * j + e] = xs[nn * KPAD + ks * 32 + kbA + e];
        }
    }

    v8f accs[nCT];

    #pragma unroll
    for (int ct = 0; ct < nCT; ++ct) {
        const int col = ct * 16 + nn;
        const float bval = bias[col];
        #pragma unroll
        for (int j = 0; j < 8; ++j) accs[ct][j] = bval;

        #pragma unroll
        for (int ks = 0; ks < nK; ++ks) {
            // B fragment (32x16 f16, col = nn). VGPR j (j<4): K = 8g+2j+{0,1};  (j>=4): K = 16+8g+2(j-4)+{0,1}.
            // Branchless: clamp row index in-bounds, select 0 past Fin.
            v16h Bv;
            #pragma unroll
            for (int j = 0; j < 8; ++j) {
                #pragma unroll
                for (int e = 0; e < 2; ++e) {
                    const int kb = ks * 32 + ((j < 4) ? (8 * g + 2 * j) : (16 + 8 * g + 2 * (j - 4))) + e;
                    const int kc = (kb < Fin) ? kb : 0;
                    const float wv = W[(unsigned)kc * F + col];
                    Bv[2 * j + e] = (kb < Fin) ? (_Float16)wv : (_Float16)0.0f;
                }
            }
            accs[ct] = __builtin_amdgcn_wmma_f32_16x16x32_f16(
                false, A[ks], false, Bv, (short)0, accs[ct], false, false);
        }
    }

    // D layout: VGPR j -> row m = j + 8*g, col = nn.
    // Hot path: full 16-row tile (always true when N % 16 == 0).
    if (row0 + 16 <= N) {
        #pragma unroll
        for (int ct = 0; ct < nCT; ++ct) {
            float* op = Out + (unsigned)(row0 + 8 * g) * F + (ct * 16 + nn);
            #pragma unroll
            for (int j = 0; j < 8; ++j) op[(unsigned)(j * F)] = accs[ct][j];
        }
    } else {
        #pragma unroll
        for (int ct = 0; ct < nCT; ++ct) {
            float* op = Out + (unsigned)(row0 + 8 * g) * F + (ct * 16 + nn);
            #pragma unroll
            for (int j = 0; j < 8; ++j)
                if (row0 + 8 * g + j < N) op[(unsigned)(j * F)] = accs[ct][j];
        }
    }
}

// ---------------- edge phase ----------------
// a[e,h] = scale * <q[dst,h,:], k[src,h,:]>; atomic-max into amax[dst,h]
template<int H, int C, int F>
__global__ void edge_score_kernel(const int* __restrict__ src, const int* __restrict__ dst,
    const float* __restrict__ Q, const float* __restrict__ Kf,
    float* __restrict__ wa, float* __restrict__ amax,
    int E, float scale)
{
    int idx = blockIdx.x * blockDim.x + threadIdx.x;
    if (idx >= E * H) return;
    int e = idx / H;
    int h = idx - e * H;
    int s = src[e], d = dst[e];
    const float* qp = Q  + (unsigned)d * F + h * C;
    const float* kp = Kf + (unsigned)s * F + h * C;
    float acc = 0.0f;
    #pragma unroll
    for (int c = 0; c < C; c += 4) {
        float4 q4 = *(const float4*)(qp + c);
        float4 k4 = *(const float4*)(kp + c);
        acc += q4.x * k4.x + q4.y * k4.y + q4.z * k4.z + q4.w * k4.w;
    }
    acc *= scale;
    wa[idx] = acc;
    atomicMaxFloat(&amax[(unsigned)d * H + h], acc);
}

// ea = exp(a - amax[dst]); denom[dst,h] += ea  (in-place on wa)
template<int H>
__global__ void edge_softmax_kernel(const int* __restrict__ dst,
    float* __restrict__ wa, const float* __restrict__ amax,
    float* __restrict__ denom, int E)
{
    int idx = blockIdx.x * blockDim.x + threadIdx.x;
    if (idx >= E * H) return;
    int e = idx / H;
    int h = idx - e * H;
    int d = dst[e];
    float m = amax[(unsigned)d * H + h];
    if (!isfinite(m)) m = 0.0f;
    float ea = expf(wa[idx] - m);
    wa[idx] = ea;
    atomicAdd(&denom[(unsigned)d * H + h], ea);
}

// agg[dst,f] += v[src,f] * ea[e,h]
template<int H, int C, int F>
__global__ void edge_scatter_kernel(const int* __restrict__ src, const int* __restrict__ dst,
    const float* __restrict__ V, const float* __restrict__ wea,
    float* __restrict__ agg, int E)
{
    unsigned idx = blockIdx.x * blockDim.x + threadIdx.x;
    unsigned total = (unsigned)E * (unsigned)F;
    if (idx >= total) return;
    unsigned e = idx / (unsigned)F;
    unsigned f = idx - e * (unsigned)F;
    int h = (int)f / C;
    int s = src[e], d = dst[e];
    float ea = wea[e * H + h];
    atomicAdd(&agg[(unsigned)d * F + f], V[(unsigned)s * F + f] * ea);
}

// x_next = agg / (denom + 1e-16) + (x @ Ws + bs)
template<int H, int C, int F>
__global__ void node_final_kernel(float* __restrict__ xout,
    const float* __restrict__ denom, const float* __restrict__ S,
    int N)
{
    int idx = blockIdx.x * blockDim.x + threadIdx.x;
    if (idx >= N * F) return;
    int n = idx / F;
    int f = idx - n * F;
    int h = f / C;
    xout[idx] = xout[idx] / (denom[(unsigned)n * H + h] + 1e-16f) + S[idx];
}

// ---------------- pooling + MLP ----------------
__global__ void pool_kernel(const float* __restrict__ x, const int* __restrict__ batch,
                            float* __restrict__ gbuf, float* __restrict__ counts, int N) {
    int idx = blockIdx.x * blockDim.x + threadIdx.x;
    if (idx >= N * 64) return;
    int n = idx >> 6, f = idx & 63;
    int b = batch[n];
    atomicAdd(&gbuf[b * 64 + f], x[idx]);
    if (f == 0) atomicAdd(&counts[b], 1.0f);
}

__global__ void pool_div_kernel(float* __restrict__ gbuf, const float* __restrict__ counts, int B) {
    int idx = blockIdx.x * blockDim.x + threadIdx.x;
    if (idx >= B * 64) return;
    gbuf[idx] /= fmaxf(counts[idx >> 6], 1.0f);
}

__global__ void mlp_kernel(const float* __restrict__ gbuf, const float* __restrict__ demo,
    const float* __restrict__ W1, const float* __restrict__ b1,
    const float* __restrict__ W2, const float* __restrict__ b2,
    float* __restrict__ out, int B)
{
    int b = blockIdx.x * blockDim.x + threadIdx.x;
    if (b >= B) return;
    float f[69];
    #pragma unroll
    for (int i = 0; i < 64; ++i) f[i] = gbuf[b * 64 + i];
    #pragma unroll
    for (int i = 0; i < 5; ++i) f[64 + i] = demo[b * 5 + i];
    float hid[32];
    for (int j = 0; j < 32; ++j) {
        float a = b1[j];
        for (int i = 0; i < 69; ++i) a += f[i] * W1[i * 32 + j];
        hid[j] = fmaxf(a, 0.0f);
    }
    for (int o = 0; o < 2; ++o) {
        float a = b2[o];
        for (int j = 0; j < 32; ++j) a += hid[j] * W2[j * 2 + o];
        out[b * 2 + o] = a;
    }
}

// ---------------- host-side per-layer driver ----------------
template<int Fin, int H, int C>
static void run_layer(float* xin, float* xout,
                      const int* src, const int* dst, int N, int E,
                      void* const* d_in, int base,
                      float* Q, float* K, float* V, float* S,
                      float* amax, float* denom, hipStream_t stream)
{
    constexpr int F  = H * C;
    constexpr int TB = 256;
    const float* Wq = (const float*)d_in[base + 0];
    const float* Wk = (const float*)d_in[base + 1];
    const float* Wv = (const float*)d_in[base + 2];
    const float* Ws = (const float*)d_in[base + 3];
    const float* bq = (const float*)d_in[base + 4];
    const float* bk = (const float*)d_in[base + 5];
    const float* bv = (const float*)d_in[base + 6];
    const float* bs = (const float*)d_in[base + 7];

    gemm4_wmma_kernel<Fin, F><<<(N + 15) / 16, 128, 0, stream>>>(
        xin, N, Wq, Wk, Wv, Ws, bq, bk, bv, bs, Q, K, V, S);

    fill_kernel<<<(N * H + TB - 1) / TB, TB, 0, stream>>>(amax, -INFINITY, N * H);
    fill_kernel<<<(N * H + TB - 1) / TB, TB, 0, stream>>>(denom, 0.0f, N * H);
    fill_kernel<<<(N * F + TB - 1) / TB, TB, 0, stream>>>(xout, 0.0f, N * F);

    // xin buffer is dead after the GEMM -> reuse as edge-score scratch [E,H]
    float* wa = xin;
    const float scale = 1.0f / sqrtf((float)C);

    edge_score_kernel<H, C, F><<<(E * H + TB - 1) / TB, TB, 0, stream>>>(
        src, dst, Q, K, wa, amax, E, scale);
    edge_softmax_kernel<H><<<(E * H + TB - 1) / TB, TB, 0, stream>>>(
        dst, wa, amax, denom, E);
    const long long totEF = (long long)E * F;
    edge_scatter_kernel<H, C, F><<<(unsigned)((totEF + TB - 1) / TB), TB, 0, stream>>>(
        src, dst, V, wa, xout, E);
    node_final_kernel<H, C, F><<<(N * F + TB - 1) / TB, TB, 0, stream>>>(
        xout, denom, S, N);
}

// ---------------- host orchestration ----------------
extern "C" void kernel_launch(void* const* d_in, const int* in_sizes, int n_in,
                              void* d_out, int out_size, void* d_ws, size_t ws_size,
                              hipStream_t stream) {
    (void)n_in; (void)out_size; (void)ws_size;
    const int*   node_idx = (const int*)d_in[0];
    const int*   ei       = (const int*)d_in[1];
    const int*   batch    = (const int*)d_in[2];
    const float* demo     = (const float*)d_in[3];
    const float* emb      = (const float*)d_in[4];

    const int N = in_sizes[0];
    const int E = in_sizes[1] / 2;
    const int B = in_sizes[3] / 5;
    const int* src = ei;
    const int* dst = ei + E;

    // workspace layout (floats); ~117 MB total
    float* p = (float*)d_ws;
    float* xa    = p; p += (size_t)N * 96;
    float* xb    = p; p += (size_t)N * 96;
    float* Q     = p; p += (size_t)N * 96;
    float* K     = p; p += (size_t)N * 96;
    float* V     = p; p += (size_t)N * 96;
    float* S     = p; p += (size_t)N * 96;
    float* amax  = p; p += (size_t)N * 3;
    float* denom = p; p += (size_t)N * 3;
    float* gbuf  = p; p += (size_t)B * 64;
    float* cnts  = p; p += B;

    const int TB = 256;
    embed_kernel<<<(N * 16 + TB - 1) / TB, TB, 0, stream>>>(node_idx, emb, xa, N);

    run_layer<16, 3, 32>(xa, xb, src, dst, N, E, d_in, 5,  Q, K, V, S, amax, denom, stream);
    run_layer<96, 1, 96>(xb, xa, src, dst, N, E, d_in, 13, Q, K, V, S, amax, denom, stream);
    run_layer<96, 1, 64>(xa, xb, src, dst, N, E, d_in, 21, Q, K, V, S, amax, denom, stream);

    // mean pool over batch (final F = 64)
    fill_kernel<<<(B * 64 + TB - 1) / TB, TB, 0, stream>>>(gbuf, 0.0f, B * 64);
    fill_kernel<<<(B + TB - 1) / TB, TB, 0, stream>>>(cnts, 0.0f, B);
    pool_kernel<<<(N * 64 + TB - 1) / TB, TB, 0, stream>>>(xb, batch, gbuf, cnts, N);
    pool_div_kernel<<<(B * 64 + TB - 1) / TB, TB, 0, stream>>>(gbuf, cnts, B);

    const float* W1 = (const float*)d_in[29];
    const float* b1 = (const float*)d_in[30];
    const float* W2 = (const float*)d_in[31];
    const float* b2 = (const float*)d_in[32];
    mlp_kernel<<<(B + 63) / 64, 64, 0, stream>>>(gbuf, demo, W1, b1, W2, b2, (float*)d_out, B);
}
